// CNN3_P_lat_3719441679190
// MI455X (gfx1250) — matrix operand order repaired and checked by reading, not compiled, not run on hardware
//
#include <hip/hip_runtime.h>
#include <hip/hip_bf16.h>
#include <hip/hip_fp16.h>
#include <stdint.h>

typedef __attribute__((ext_vector_type(16))) _Float16 v16h;
typedef __attribute__((ext_vector_type(8)))  _Float16 v8h;
typedef __attribute__((ext_vector_type(8)))  float    v8f;

// ---- problem constants ----
#define NB    4096          // batch
#define CLn   128           // context length
#define ILn   64            // inst length
#define PCc   64
#define CH1c  128
#define CH2c  256
#define CH3c  256
#define L0c   127
#define L1c   125
#define L2c   123
#define L3c   121
#define C3OUT (CH3c * L3c)      // 30976
#define F1c   400
#define F1P   448               // padded to /64 (block M tile)
#define F1IN  39168             // C3OUT + 64*128
#define XTN   (ILn * CLn)       // 8192

#define KSTEP 64
#define LDP   72                // padded K stride in LDS (144B = 9*16B)

// ---------------------------------------------------------------------------
// prep kernels (f32 -> f16 conversions / transposes / padding)
// ---------------------------------------------------------------------------
__global__ __launch_bounds__(256) void prep_xt(const float* __restrict__ x,
                                               _Float16* __restrict__ xth) {
  int t = blockIdx.x * 256 + threadIdx.x;          // t = b*8192 + c*128 + i
  if (t >= NB * XTN) return;
  int b = t >> 13, rem = t & 8191;
  int c = rem >> 7, i = rem & 127;
  xth[t] = (_Float16)x[(size_t)b * XTN + i * ILn + c];
}

__global__ __launch_bounds__(256) void prep_wp(const float* __restrict__ Wp,
                                               _Float16* __restrict__ wph) {
  int t = blockIdx.x * 256 + threadIdx.x;          // t = p*64 + c
  if (t >= PCc * ILn) return;
  int p = t >> 6, c = t & 63;
  wph[t] = (_Float16)Wp[p * (ILn * 2) + c * 2 + 1];   // Wp[:,:,1]
}

__global__ __launch_bounds__(256) void prep_convw(const float* __restrict__ W,
                                                  _Float16* __restrict__ Wh,
                                                  int Cout, int Cin) {
  int t = blockIdx.x * 256 + threadIdx.x;          // t = o*Cin*4 + c*4 + k
  int total = Cout * Cin * 4;
  if (t >= total) return;
  int o = t / (Cin * 4), rem = t % (Cin * 4);
  int c = rem >> 2, k = rem & 3;
  Wh[t] = (k < 3) ? (_Float16)W[((size_t)o * Cin + c) * 3 + k] : (_Float16)0.f;
}

__global__ __launch_bounds__(256) void prep_fc1w(const float* __restrict__ Wf1,
                                                 _Float16* __restrict__ Wh) {
  int t = blockIdx.x * 256 + threadIdx.x;          // t = f*39168 + j
  if (t >= F1P * F1IN) return;
  int f = t / F1IN, j = t % F1IN;
  Wh[t] = (f < F1c) ? (_Float16)Wf1[(size_t)f * F1IN + j] : (_Float16)0.f;
}

// base[b,p] = sum_c Wp[p,c,0] * x[b, c]   (xt[:,:,0] == x[b, 0*64+c])
__global__ __launch_bounds__(256) void convp_base(const float* __restrict__ x,
                                                  const float* __restrict__ Wp,
                                                  float* __restrict__ baseT) {
  int t = blockIdx.x * 256 + threadIdx.x;
  if (t >= NB * PCc) return;
  int b = t >> 6, p = t & 63;
  float s = 0.f;
  #pragma unroll 8
  for (int c = 0; c < ILn; ++c)
    s += Wp[p * (ILn * 2) + c * 2] * x[(size_t)b * XTN + c];
  baseT[t] = s;
}

// ---------------------------------------------------------------------------
// Generic WMMA GEMM: D[m,n] = sum_k A[m,k] * B[k,n]  (+ bias / base, ReLU)
//   MODE 0: convp    B[c, b*127+i]   = xth[b, c, i+1]
//   MODE 1: conv k=3 B[c*4+k, b*L+l] = in[b, c, l+k]   (k==3 -> 0 pad)
//   MODE 2: fc1      B[j, b]         = j<30976 ? h3[b,j] : xth[b, j-30976]
// Block: 256 thr = 8 waves; tile 64M x 256N; wave tile 16M x 128N; K step 64.
// Double-buffered LDS, ONE barrier per K step; A tile staged with
// GLOBAL_LOAD_ASYNC_TO_LDS_B128 (ASYNCcnt).
// ---------------------------------------------------------------------------
template <int MODE>
__global__ __launch_bounds__(256) void gemm_wmma(
    const _Float16* __restrict__ A,    // [Mpad][Kpad] f16
    const _Float16* __restrict__ S,    // layer input (f16)
    const _Float16* __restrict__ S2,   // fc1: xth
    const float* __restrict__ bias,
    const float* __restrict__ baseT,   // convp shared term
    _Float16* __restrict__ outH,       // conv outputs
    float* __restrict__ outF,          // fc1 output
    int Mpad, int Kpad, int Mreal, int Cin, int Lin, int Lout) {
  __shared__ _Float16 sA[2][64 * LDP];    // [m][k]  2 x 9 KB
  __shared__ _Float16 sB[2][256 * LDP];   // [n][k]  2 x 36 KB

  const int tid   = threadIdx.x;
  const int lane  = tid & 31;
  const int wave  = tid >> 5;
  const int waveM = wave >> 1;         // 0..3  -> 16 M rows each
  const int waveN = wave & 1;          // 0..1  -> 128 N cols each
  const int m0    = blockIdx.y * 64;
  const int n0    = blockIdx.x * 256;

  v8f acc[8] = {v8f{}, v8f{}, v8f{}, v8f{}, v8f{}, v8f{}, v8f{}, v8f{}};

  // column this thread stages
  const int nst = n0 + tid;
  int bb = 0, ll = 0;
  if constexpr (MODE == 0) { bb = nst / L0c; ll = nst % L0c; }
  if constexpr (MODE == 1) { bb = nst / Lout; ll = nst % Lout; }

  // per-thread A staging slot: 16 f16 (2 x b128) of the 64x64 tile
  const int am = tid >> 2;             // row 0..63
  const int ak = (tid & 3) * 16;       // col 0,16,32,48

  // issue async global->LDS copies for the A tile of this K step
  auto stageA = [&](int buf, int k0) {
    const _Float16* ga = &A[(size_t)(m0 + am) * Kpad + k0 + ak];
    unsigned l0 = (unsigned)(uintptr_t)(&sA[buf][am * LDP + ak]);
    asm volatile("global_load_async_to_lds_b128 %0, %1, off"
                 :: "v"(l0), "v"(ga) : "memory");
    unsigned l1 = l0 + 16;
    const _Float16* ga1 = ga + 8;
    asm volatile("global_load_async_to_lds_b128 %0, %1, off"
                 :: "v"(l1), "v"(ga1) : "memory");
  };

  // gather one B column (64 f16 along K) into LDS
  auto stageB = [&](int buf, int k0) {
    _Float16* dst = &sB[buf][tid * LDP];
    if constexpr (MODE == 0) {
      const _Float16* src = &S[(size_t)bb * XTN + (ll + 1)];
      #pragma unroll
      for (int r = 0; r < KSTEP; ++r)
        dst[r] = src[(k0 + r) * CLn];
    } else if constexpr (MODE == 1) {
      const _Float16* src = &S[((size_t)bb * Cin) * Lin + ll];
      #pragma unroll
      for (int r = 0; r < KSTEP; ++r) {
        int rr = k0 + r;
        int c = rr >> 2, k = rr & 3;
        _Float16 v = src[c * Lin + (k < 3 ? k : 2)];
        dst[r] = (k < 3) ? v : (_Float16)0.f;
      }
    } else {
      #pragma unroll
      for (int r = 0; r < KSTEP; ++r) {
        int j = k0 + r;
        dst[r] = (j < C3OUT) ? S[(size_t)nst * C3OUT + j]
                             : S2[(size_t)nst * XTN + (j - C3OUT)];
      }
    }
  };

  const int arow = waveM * 16 + (lane & 15);
  const int akf  = (lane >> 4) * 8;     // A frag K half offset
  const int bn   = waveN * 128 + (lane & 15);
  const int bk   = (lane >> 4) * 8;     // B frag K half offset

  // prologue: stage first K block into buffer 0
  stageA(0, 0);
  stageB(0, 0);
  int buf = 0;

  for (int k0 = 0; k0 < Kpad; k0 += KSTEP) {
    asm volatile("s_wait_asynccnt 0x0" ::: "memory");
    __syncthreads();                    // tile `buf` complete in LDS.
    // This is the ONLY barrier per K step: the next write to `buf` happens
    // after the *next* iteration's barrier, which every wave reaches only
    // after finishing this iteration's reads (dscnt drained before signal).
    if (k0 + KSTEP < Kpad) {            // prefetch next tile into other buffer
      stageA(buf ^ 1, k0 + KSTEP);
      stageB(buf ^ 1, k0 + KSTEP);
    }
    // ---- compute on tile `buf`: 2 sub-steps of K=32, 8 WMMA each ----
    #pragma unroll
    for (int kk = 0; kk < 2; ++kk) {
      const v8h* pA = (const v8h*)&sA[buf][arow * LDP + kk * 32 + akf];
      v8h alo = pA[0];                  // K = +0..7
      v8h ahi = pA[2];                  // K = +16..23
      v16h af = __builtin_shufflevector(alo, ahi, 0, 1, 2, 3, 4, 5, 6, 7,
                                        8, 9, 10, 11, 12, 13, 14, 15);
      v16h bfr[8];
      #pragma unroll
      for (int s = 0; s < 8; ++s) {
        const v8h* pB =
            (const v8h*)&sB[buf][(bn + s * 16) * LDP + kk * 32 + bk];
        v8h blo = pB[0], bhi = pB[2];
        bfr[s] = __builtin_shufflevector(blo, bhi, 0, 1, 2, 3, 4, 5, 6, 7,
                                         8, 9, 10, 11, 12, 13, 14, 15);
      }
      #pragma unroll
      for (int s = 0; s < 8; ++s)
        acc[s] = __builtin_amdgcn_wmma_f32_16x16x32_f16(
            false, af, false, bfr[s], (short)0, acc[s], false, false);
    }
    buf ^= 1;
  }

  // ---- epilogue: VGPR r -> M = r + 8*(lane>=16); N = lane&15 ----
  const int mbase = m0 + waveM * 16 + (lane >> 4) * 8;
  const int nbase = n0 + waveN * 128 + (lane & 15);
  #pragma unroll
  for (int s = 0; s < 8; ++s) {
    int n = nbase + s * 16;
    #pragma unroll
    for (int r = 0; r < 8; ++r) {
      int m = mbase + r;
      if (m < Mreal) {
        float v = acc[s][r];
        if constexpr (MODE == 0) {
          int b = n / L0c, i = n % L0c;
          v += baseT[b * PCc + m] + bias[m];
          v = fmaxf(v, 0.f);
          outH[((size_t)b * PCc + m) * L0c + i] = (_Float16)v;
        } else if constexpr (MODE == 1) {
          int b = n / Lout, l = n % Lout;
          v = fmaxf(v + bias[m], 0.f);
          outH[((size_t)b * Mpad + m) * Lout + l] = (_Float16)v;
        } else {
          v = fmaxf(v + bias[m], 0.f);
          outF[(size_t)n * F1c + m] = v;
        }
      }
    }
  }
}

// ---------------------------------------------------------------------------
__global__ __launch_bounds__(256) void fc2_k(const float* __restrict__ fc1,
                                             const float* __restrict__ Wf2,
                                             const float* __restrict__ bf2,
                                             float* __restrict__ out) {
  int b = blockIdx.x * 256 + threadIdx.x;
  if (b >= NB) return;
  float s = 0.f;
  #pragma unroll 8
  for (int f = 0; f < F1c; ++f) s += fc1[(size_t)b * F1c + f] * Wf2[f];
  out[b] = s + bf2[0];
}

// ---------------------------------------------------------------------------
static inline int cdiv(long long a, long long b) { return (int)((a + b - 1) / b); }

extern "C" void kernel_launch(void* const* d_in, const int* in_sizes, int n_in,
                              void* d_out, int out_size, void* d_ws, size_t ws_size,
                              hipStream_t stream) {
  const float* x   = (const float*)d_in[0];
  const float* Wp  = (const float*)d_in[1];
  const float* bp  = (const float*)d_in[2];
  const float* W1  = (const float*)d_in[3];
  const float* b1  = (const float*)d_in[4];
  const float* W2  = (const float*)d_in[5];
  const float* b2  = (const float*)d_in[6];
  const float* W3  = (const float*)d_in[7];
  const float* b3  = (const float*)d_in[8];
  const float* Wf1 = (const float*)d_in[9];
  const float* bf1 = (const float*)d_in[10];
  const float* Wf2 = (const float*)d_in[11];
  const float* bf2 = (const float*)d_in[12];
  float* out = (float*)d_out;

  // ---- carve workspace ----
  char* p = (char*)d_ws;
  auto a16 = [&](size_t n) {
    _Float16* r = (_Float16*)p;
    p += ((n * 2 + 255) / 256) * 256;
    return r;
  };
  auto a32 = [&](size_t n) {
    float* r = (float*)p;
    p += ((n * 4 + 255) / 256) * 256;
    return r;
  };
  _Float16* xth  = a16((size_t)NB * XTN);
  _Float16* wph  = a16((size_t)PCc * ILn);
  _Float16* w1h  = a16((size_t)CH1c * PCc * 4);
  _Float16* w2h  = a16((size_t)CH2c * CH1c * 4);
  _Float16* w3h  = a16((size_t)CH3c * CH2c * 4);
  _Float16* wf1h = a16((size_t)F1P * F1IN);
  _Float16* h0   = a16((size_t)NB * PCc * L0c);
  _Float16* h1   = a16((size_t)NB * CH1c * L1c);
  _Float16* h2   = a16((size_t)NB * CH2c * L2c);
  _Float16* h3   = a16((size_t)NB * CH3c * L3c);
  float* baseT = a32((size_t)NB * PCc);
  float* fc1o  = a32((size_t)NB * F1c);

  // ---- prep ----
  prep_xt<<<cdiv((long long)NB * XTN, 256), 256, 0, stream>>>(x, xth);
  prep_wp<<<cdiv(PCc * ILn, 256), 256, 0, stream>>>(Wp, wph);
  prep_convw<<<cdiv(CH1c * PCc * 4, 256), 256, 0, stream>>>(W1, w1h, CH1c, PCc);
  prep_convw<<<cdiv(CH2c * CH1c * 4, 256), 256, 0, stream>>>(W2, w2h, CH2c, CH1c);
  prep_convw<<<cdiv(CH3c * CH2c * 4, 256), 256, 0, stream>>>(W3, w3h, CH3c, CH2c);
  prep_fc1w<<<cdiv((long long)F1P * F1IN, 256), 256, 0, stream>>>(Wf1, wf1h);
  convp_base<<<cdiv(NB * PCc, 256), 256, 0, stream>>>(x, Wp, baseT);

  // ---- GEMM chain ----
  // convp: M=64, N=4096*127, K=64
  gemm_wmma<0><<<dim3((NB * L0c) / 256, PCc / 64), 256, 0, stream>>>(
      wph, xth, nullptr, bp, baseT, h0, nullptr, PCc, ILn, PCc, ILn, CLn, L0c);
  // conv1: M=128, N=4096*125, K=256
  gemm_wmma<1><<<dim3((NB * L1c) / 256, CH1c / 64), 256, 0, stream>>>(
      w1h, h0, nullptr, b1, nullptr, h1, nullptr, CH1c, PCc * 4, CH1c, PCc, L0c, L1c);
  // conv2: M=256, N=4096*123, K=512
  gemm_wmma<1><<<dim3((NB * L2c) / 256, CH2c / 64), 256, 0, stream>>>(
      w2h, h1, nullptr, b2, nullptr, h2, nullptr, CH2c, CH1c * 4, CH2c, CH1c, L1c, L2c);
  // conv3: M=256, N=4096*121, K=1024
  gemm_wmma<1><<<dim3((NB * L3c) / 256, CH3c / 64), 256, 0, stream>>>(
      w3h, h2, nullptr, b3, nullptr, h3, nullptr, CH3c, CH2c * 4, CH3c, CH2c, L2c, L3c);
  // fc1: M=448(pad of 400), N=4096, K=39168
  gemm_wmma<2><<<dim3(NB / 256, F1P / 64), 256, 0, stream>>>(
      wf1h, h3, xth, bf1, nullptr, nullptr, fc1o, F1P, F1IN, F1c, 0, 0, 0);

  // ---- fc2 ----
  fc2_k<<<cdiv(NB, 256), 256, 0, stream>>>(fc1o, Wf2, bf2, out);
}
